// SwinBlock_tp_56006373539972
// MI455X (gfx1250) — compile-verified, matrix operand order checked by reading, not run on hardware
//
#include <hip/hip_runtime.h>
#include <hip/hip_bf16.h>
#include <math.h>

// ---------------------------------------------------------------------------
// Swin block for MI455X (gfx1250, wave32, WMMA). All GEMMs run through
// v_wmma_f32_16x16x32_bf16 with fp32 accumulation; elementwise work is fused
// into epilogues. GEMM tile staging uses CDNA5 async global->LDS copies
// (global_load_async_to_lds_b128, ASYNCcnt) so the double-buffer loop does
// no VGPR round-trip. ~840 GFLOP total -> matrix-unit bound, bf16 path.
// ---------------------------------------------------------------------------

typedef __bf16  bf16_t;
typedef __bf16  v16bf __attribute__((ext_vector_type(16)));
typedef float   v8f   __attribute__((ext_vector_type(8)));

#define SB_H      64
#define SB_WS     8
#define SB_SHIFT  4
#define SB_C      512
#define SB_NH     16
#define SB_DH     32
#define SB_B      32
#define SB_SW     64           // WS*WS tokens per window (== attn seq len here)
#define SB_NB     2048         // window-batch
#define SB_ROWS   131072       // total tokens = 4096*32 = SW*NB
#define FC_CHUNK  16384        // rows per FC1/FC2 chunk

// ---------------- WMMA helpers -------------------------------------------

__device__ __forceinline__ v8f wmma_bf16(v16bf a, v16bf b, v8f c) {
  // D = A(16x32) * B(32x16) + C, fp32 accum
  return __builtin_amdgcn_wmma_f32_16x16x32_bf16(
      /*neg_a=*/false, a, /*neg_b=*/false, b,
      /*c_mod=*/(short)0, c, /*reuse_a=*/false, /*reuse_b=*/false);
}

// Load one 16x32 bf16 fragment from an LDS tile (row-major, ldk elements).
// Per ISA 16-bit A/B layout: lane l -> row (l&15); element i -> K =
// (i>>3)*16 + 8*(l>>4) + (i&7).  => two contiguous 16B LDS reads per lane.
__device__ __forceinline__ v16bf ld_frag(const bf16_t* t, int row0, int ldk, int k0) {
  const int l = threadIdx.x & 31;
  const int m = l & 15;
  const int h = l >> 4;
  const bf16_t* p = t + (row0 + m) * ldk + k0 + 8 * h;
  union U { v16bf v; struct { int4 a, b; } q; } u;
  u.q.a = *reinterpret_cast<const int4*>(p);
  u.q.b = *reinterpret_cast<const int4*>(p + 16);
  return u.v;
}

// ---------------- CDNA5 async global->LDS staging --------------------------

// Low 32 bits of a generic pointer to a __shared__ object are the LDS byte
// address (ISA 10.2).
__device__ __forceinline__ unsigned lds_byte_off(const void* p) {
  return (unsigned)(uintptr_t)p;
}

// One 16B async copy global->LDS per call (GLOBAL_LOAD_ASYNC_TO_LDS_B128,
// GV addressing, tracked with ASYNCcnt).
__device__ __forceinline__ void async_copy_b128(bf16_t* lds_dst, const bf16_t* gsrc) {
  unsigned dst = lds_byte_off(lds_dst);
  unsigned long long src = (unsigned long long)(uintptr_t)gsrc;
  asm volatile("global_load_async_to_lds_b128 %0, %1, off"
               :: "v"(dst), "v"(src) : "memory");
}

__device__ __forceinline__ void wait_asynccnt0() {
  asm volatile("s_wait_asynccnt 0" ::: "memory");
}

// ---------------- Generic bf16 GEMM: C[M,N] = A[M,K] * Bt[N,K]^T ----------
// Block: 256 threads (8 waves). Block tile 128x128, wave tile 64x32,
// K-step 32 (one WMMA), double-buffered LDS, async tile staging.

__device__ __forceinline__ void stage_tile_async(bf16_t* dst, const bf16_t* src, size_t ld) {
  // copy 128 rows x 32 bf16 (row-major) into dst[128*32]; 512 x 16B chunks
  const int t = threadIdx.x;
#pragma unroll
  for (int c = 0; c < 2; ++c) {
    int chunk = t + c * 256;
    int row = chunk >> 2, seg = chunk & 3;
    async_copy_b128(dst + row * 32 + seg * 8, src + (size_t)row * ld + seg * 8);
  }
}

template <class Epi>
__global__ __launch_bounds__(256) void gemm_bf16_kernel(
    const bf16_t* __restrict__ A, const bf16_t* __restrict__ Bt,
    int M, int N, int K, Epi epi) {
  __shared__ __align__(16) bf16_t sA[2][128 * 32];
  __shared__ __align__(16) bf16_t sB[2][128 * 32];

  const int bm = blockIdx.y * 128;
  const int bn = blockIdx.x * 128;
  const int wave = threadIdx.x >> 5;
  const int wm = (wave >> 2) * 64;   // 0 / 64
  const int wn = (wave & 3) * 32;    // 0,32,64,96

  v8f acc[4][2];
#pragma unroll
  for (int mi = 0; mi < 4; ++mi)
#pragma unroll
    for (int ni = 0; ni < 2; ++ni)
      acc[mi][ni] = v8f{0.f, 0.f, 0.f, 0.f, 0.f, 0.f, 0.f, 0.f};

  const bf16_t* Ab = A + (size_t)bm * K;
  const bf16_t* Bb = Bt + (size_t)bn * K;
  const int nk = K >> 5;

  stage_tile_async(sA[0], Ab, K);
  stage_tile_async(sB[0], Bb, K);
  wait_asynccnt0();
  __syncthreads();

  for (int kt = 0; kt < nk; ++kt) {
    const int buf = kt & 1;
    if (kt + 1 < nk) {
      stage_tile_async(sA[buf ^ 1], Ab + (size_t)(kt + 1) * 32, K);
      stage_tile_async(sB[buf ^ 1], Bb + (size_t)(kt + 1) * 32, K);
    }
    if (kt + 2 < nk) {  // software prefetch of the k+2 tiles (global_prefetch_b8)
      __builtin_prefetch(Ab + (size_t)(threadIdx.x >> 1) * K + (kt + 2) * 32, 0, 1);
      __builtin_prefetch(Bb + (size_t)(threadIdx.x >> 1) * K + (kt + 2) * 32, 0, 1);
    }
    v16bf af[4], bfr[2];
#pragma unroll
    for (int mi = 0; mi < 4; ++mi) af[mi] = ld_frag(sA[buf], wm + mi * 16, 32, 0);
#pragma unroll
    for (int ni = 0; ni < 2; ++ni) bfr[ni] = ld_frag(sB[buf], wn + ni * 16, 32, 0);
#pragma unroll
    for (int mi = 0; mi < 4; ++mi)
#pragma unroll
      for (int ni = 0; ni < 2; ++ni)
        acc[mi][ni] = wmma_bf16(af[mi], bfr[ni], acc[mi][ni]);
    wait_asynccnt0();     // next-tile async copies landed in LDS
    __syncthreads();
  }

  // Epilogue. D layout: VGPR r -> row r + 8*(lane>>4); col = lane&15.
  const int l = threadIdx.x & 31;
  const int cn = l & 15, hf = l >> 4;
#pragma unroll
  for (int mi = 0; mi < 4; ++mi)
#pragma unroll
    for (int ni = 0; ni < 2; ++ni)
#pragma unroll
      for (int r = 0; r < 8; ++r) {
        int row = bm + wm + mi * 16 + hf * 8 + r;
        int col = bn + wn + ni * 16 + cn;
        epi(row, col, acc[mi][ni][r]);
      }
}

// ---------------- Epilogue functors ---------------------------------------

struct EpiQKV {           // qkv = x@Wqkv + b  (bf16 out, [rows,1536])
  const float* bias; bf16_t* out;
  __device__ void operator()(int r, int c, float v) const {
    out[(size_t)r * 1536 + c] = (bf16_t)(v + bias[c]);
  }
};

struct EpiProj {          // hidden = shortcut + unshift(reverse(ctx@Wp + b + xw))
  const float* bias; const bf16_t* xw; const float* shortcut; float* hidden;
  __device__ void operator()(int r, int c, float v) const {
    int s = r >> 11, n = r & 2047;            // r = s*NB + n
    int hn = s >> 3, wn = s & 7;
    int intra = n >> 5, b = n & 31;
    int i = ((hn << 3) + (intra >> 3) + SB_SHIFT) & 63;   // roll(+SHIFT)
    int j = ((wn << 3) + (intra & 7) + SB_SHIFT) & 63;
    size_t flat = (((size_t)(i * 64 + j)) * SB_B + b) * SB_C + c;
    hidden[flat] = v + bias[c] + (float)xw[(size_t)r * SB_C + c] + shortcut[flat];
  }
};

struct EpiGelu {          // h = gelu(z@W1 + b)  (bf16 out, [chunk,2048])
  const float* bias; bf16_t* out;
  __device__ void operator()(int r, int c, float v) const {
    float x = v + bias[c];
    float g = 0.5f * x * (1.0f + erff(x * 0.70710678118654752f));
    out[(size_t)r * 2048 + c] = (bf16_t)g;
  }
};

struct EpiFC2 {           // out = hidden + (h@W2 + b + ln2out)
  const float* bias; const bf16_t* ln2; float* io; int row0;
  __device__ void operator()(int r, int c, float v) const {
    size_t g = (size_t)(row0 + r) * SB_C + c;
    io[g] = io[g] + v + bias[c] + (float)ln2[g];
  }
};

// ---------------- LayerNorm kernels (1 wave per token, wave32) ------------

__device__ __forceinline__ void wave_ln_stats(const float* row, int lane,
                                              float vals[16], float& mu, float& rstd) {
  float s = 0.f;
#pragma unroll
  for (int j = 0; j < 16; ++j) { vals[j] = row[lane * 16 + j]; s += vals[j]; }
#pragma unroll
  for (int m = 1; m < 32; m <<= 1) s += __shfl_xor(s, m, 32);
  mu = s * (1.0f / 512.0f);
  float q = 0.f;
#pragma unroll
  for (int j = 0; j < 16; ++j) { float d = vals[j] - mu; q += d * d; }
#pragma unroll
  for (int m = 1; m < 32; m <<= 1) q += __shfl_xor(q, m, 32);
  rstd = rsqrtf(q * (1.0f / 512.0f) + 1e-5f);
}

// LN1 + roll(-SHIFT) + window partition, fp32 -> bf16 window layout
__global__ __launch_bounds__(256) void ln1_window_kernel(
    const float* __restrict__ x, const float* __restrict__ g,
    const float* __restrict__ b, bf16_t* __restrict__ xw) {
  const int tk = blockIdx.x * 8 + (threadIdx.x >> 5);   // token 0..131071
  const int lane = threadIdx.x & 31;
  float vals[16], mu, rstd;
  wave_ln_stats(x + (size_t)tk * SB_C, lane, vals, mu, rstd);

  // token (p,b) -> window row r = s*NB + n after roll(-SHIFT)
  const int p = tk >> 5, bidx = tk & 31;
  const int hh = p >> 6, ww = p & 63;
  const int i = (hh + 64 - SB_SHIFT) & 63;
  const int j = (ww + 64 - SB_SHIFT) & 63;
  const int s = ((i >> 3) << 3) + (j >> 3);
  const int n = (((i & 7) << 3) + (j & 7)) * SB_B + bidx;
  bf16_t* orow = xw + ((size_t)s * SB_NB + n) * SB_C;
#pragma unroll
  for (int jj = 0; jj < 16; ++jj) {
    int c = lane * 16 + jj;
    orow[c] = (bf16_t)((vals[jj] - mu) * rstd * g[c] + b[c]);
  }
}

// LN2: fp32 hidden -> bf16, same row order
__global__ __launch_bounds__(256) void ln2_kernel(
    const float* __restrict__ x, const float* __restrict__ g,
    const float* __restrict__ b, bf16_t* __restrict__ out) {
  const int tk = blockIdx.x * 8 + (threadIdx.x >> 5);
  const int lane = threadIdx.x & 31;
  float vals[16], mu, rstd;
  wave_ln_stats(x + (size_t)tk * SB_C, lane, vals, mu, rstd);
  bf16_t* orow = out + (size_t)tk * SB_C;
#pragma unroll
  for (int jj = 0; jj < 16; ++jj) {
    int c = lane * 16 + jj;
    orow[c] = (bf16_t)((vals[jj] - mu) * rstd * g[c] + b[c]);
  }
}

// ---------------- Windowed attention (1 wave per (n, head)) ----------------

__device__ __forceinline__ int sb_rgn(int c) { return c < 56 ? 0 : (c < 60 ? 1 : 2); }
__device__ __forceinline__ int sb_region(int wid, int p) {
  int rc = ((wid >> 3) << 3) + (p >> 3);
  int cc = ((wid & 7) << 3) + (p & 7);
  return sb_rgn(rc) * 3 + sb_rgn(cc);
}

__global__ __launch_bounds__(128) void attn_kernel(
    const bf16_t* __restrict__ qkv, bf16_t* __restrict__ ctx) {
  // per wave: Q[64][32] | K[64][32] | Vt[32][64]  (12KB); P reuses Q+K (8KB)
  __shared__ __align__(16) bf16_t smem[4 * 6144];
  const int wave = threadIdx.x >> 5, lane = threadIdx.x & 31;
  const int unit = blockIdx.x * 4 + wave;     // 0..32767
  const int n = unit >> 4, h = unit & 15;

  bf16_t* Q  = smem + wave * 6144;
  bf16_t* Kl = Q + 2048;
  bf16_t* Vt = Kl + 2048;
  bf16_t* P  = Q;                              // reuse after QK^T

  // stage Q,K row-major (async 16B copies); V transposed via VGPRs.
  // qkv row = t*NB + n, cols [3C].
#pragma unroll
  for (int rr = 0; rr < 2; ++rr) {
    const int t = lane + rr * 32;
    const size_t base = ((size_t)t * SB_NB + n) * (3 * SB_C) + h * SB_DH;
#pragma unroll
    for (int u = 0; u < 4; ++u) {
      async_copy_b128(Q + t * 32 + u * 8, qkv + base + u * 8);
      async_copy_b128(Kl + t * 32 + u * 8, qkv + base + SB_C + u * 8);
    }
    const bf16_t* vs = qkv + base + 2 * SB_C;
#pragma unroll
    for (int d = 0; d < 32; ++d) Vt[d * 64 + t] = vs[d];
  }
  wait_asynccnt0();
  __syncthreads();

  // scores = Q K^T   (16 wmma, K=32 in one shot)
  v8f sc[4][4];
#pragma unroll
  for (int mi = 0; mi < 4; ++mi)
#pragma unroll
    for (int ti = 0; ti < 4; ++ti)
      sc[mi][ti] = v8f{0.f, 0.f, 0.f, 0.f, 0.f, 0.f, 0.f, 0.f};
  {
    v16bf aq[4], bk[4];
#pragma unroll
    for (int mi = 0; mi < 4; ++mi) aq[mi] = ld_frag(Q, mi * 16, 32, 0);
#pragma unroll
    for (int ti = 0; ti < 4; ++ti) bk[ti] = ld_frag(Kl, ti * 16, 32, 0);
#pragma unroll
    for (int mi = 0; mi < 4; ++mi)
#pragma unroll
      for (int ti = 0; ti < 4; ++ti)
        sc[mi][ti] = wmma_bf16(aq[mi], bk[ti], sc[mi][ti]);
  }

  // mask + softmax (row reduce: 4 regs + shfl_xor over the 16-lane half)
  const float scale = 0.17677669529663687f;   // 1/sqrt(32)
  const int wid = n & 63;
  const int cn = lane & 15, hf = lane >> 4;
#pragma unroll
  for (int mi = 0; mi < 4; ++mi) {
#pragma unroll
    for (int r = 0; r < 8; ++r) {
      const int srow = mi * 16 + hf * 8 + r;
      const int rs = sb_region(wid, srow);
      float v[4]; float mx = -3.0e30f;
#pragma unroll
      for (int ti = 0; ti < 4; ++ti) {
        const int t = ti * 16 + cn;
        float x = sc[mi][ti][r] * scale;
        if (rs != sb_region(wid, t)) x = -10000.0f;
        v[ti] = x; mx = fmaxf(mx, x);
      }
#pragma unroll
      for (int m = 1; m < 16; m <<= 1) mx = fmaxf(mx, __shfl_xor(mx, m, 32));
      float sum = 0.f;
#pragma unroll
      for (int ti = 0; ti < 4; ++ti) { v[ti] = __expf(v[ti] - mx); sum += v[ti]; }
#pragma unroll
      for (int m = 1; m < 16; m <<= 1) sum += __shfl_xor(sum, m, 32);
      const float inv = 1.0f / sum;
#pragma unroll
      for (int ti = 0; ti < 4; ++ti)
        P[srow * 64 + ti * 16 + cn] = (bf16_t)(v[ti] * inv);
    }
  }
  __syncthreads();

  // ctx = P V   (16 wmma: 4 s-tiles x 2 d-tiles x 2 k-steps)
  v8f ov[4][2];
#pragma unroll
  for (int mi = 0; mi < 4; ++mi)
#pragma unroll
    for (int ni = 0; ni < 2; ++ni)
      ov[mi][ni] = v8f{0.f, 0.f, 0.f, 0.f, 0.f, 0.f, 0.f, 0.f};
#pragma unroll
  for (int kk = 0; kk < 2; ++kk) {
    v16bf ap[4], bv[2];
#pragma unroll
    for (int mi = 0; mi < 4; ++mi) ap[mi] = ld_frag(P, mi * 16, 64, kk * 32);
#pragma unroll
    for (int ni = 0; ni < 2; ++ni) bv[ni] = ld_frag(Vt, ni * 16, 64, kk * 32);
#pragma unroll
    for (int mi = 0; mi < 4; ++mi)
#pragma unroll
      for (int ni = 0; ni < 2; ++ni)
        ov[mi][ni] = wmma_bf16(ap[mi], bv[ni], ov[mi][ni]);
  }

  // scatter ctx -> [(s*NB+n), C] bf16
#pragma unroll
  for (int mi = 0; mi < 4; ++mi)
#pragma unroll
    for (int ni = 0; ni < 2; ++ni)
#pragma unroll
      for (int r = 0; r < 8; ++r) {
        const int srow = mi * 16 + hf * 8 + r;
        const int d = ni * 16 + cn;
        ctx[((size_t)srow * SB_NB + n) * SB_C + h * SB_DH + d] = (bf16_t)ov[mi][ni][r];
      }
}

// ---------------- weight convert+transpose: W[K,N] f32 -> Wt[N,K] bf16 ----

__global__ __launch_bounds__(256) void wconv_kernel(
    const float* __restrict__ w, bf16_t* __restrict__ wt, int K, int N) {
  int idx = blockIdx.x * 256 + threadIdx.x;
  if (idx >= K * N) return;
  int nn = idx / K, kk = idx - nn * K;        // wt row-major [N][K]
  wt[idx] = (bf16_t)w[(size_t)kk * N + nn];
}

// ---------------------------------------------------------------------------

extern "C" void kernel_launch(void* const* d_in, const int* in_sizes, int n_in,
                              void* d_out, int out_size, void* d_ws, size_t ws_size,
                              hipStream_t stream) {
  (void)in_sizes; (void)n_in; (void)out_size; (void)ws_size;

  const float* hs     = (const float*)d_in[0];
  const float* ln1_g  = (const float*)d_in[1];
  const float* ln1_b  = (const float*)d_in[2];
  const float* w_qkv  = (const float*)d_in[3];
  const float* b_qkv  = (const float*)d_in[4];
  const float* w_proj = (const float*)d_in[5];
  const float* b_proj = (const float*)d_in[6];
  const float* ln2_g  = (const float*)d_in[7];
  const float* ln2_b  = (const float*)d_in[8];
  const float* w_fc1  = (const float*)d_in[9];
  const float* b_fc1  = (const float*)d_in[10];
  const float* w_fc2  = (const float*)d_in[11];
  const float* b_fc2  = (const float*)d_in[12];
  float* out = (float*)d_out;                  // also used as `hidden` buffer

  // workspace carve-up
  char* ws = (char*)d_ws;
  size_t o = 0;
  auto carve = [&](size_t bytes) { char* p = ws + o; o += (bytes + 255) & ~(size_t)255; return p; };
  bf16_t* xw     = (bf16_t*)carve((size_t)SB_ROWS * SB_C * 2);        // 128MB
  bf16_t* qkvb   = (bf16_t*)carve((size_t)SB_ROWS * 3 * SB_C * 2);    // 384MB
  bf16_t* ctxb   = (bf16_t*)carve((size_t)SB_ROWS * SB_C * 2);        // 128MB
  bf16_t* ln2buf = (bf16_t*)carve((size_t)SB_ROWS * SB_C * 2);        // 128MB
  bf16_t* fc1buf = (bf16_t*)carve((size_t)FC_CHUNK * 2048 * 2);       //  64MB
  bf16_t* wqkvT  = (bf16_t*)carve((size_t)1536 * 512 * 2);
  bf16_t* wprojT = (bf16_t*)carve((size_t)512 * 512 * 2);
  bf16_t* wfc1T  = (bf16_t*)carve((size_t)2048 * 512 * 2);
  bf16_t* wfc2T  = (bf16_t*)carve((size_t)512 * 2048 * 2);

  // 0) weights -> bf16 transposed [N][K]
  wconv_kernel<<<(512 * 1536 + 255) / 256, 256, 0, stream>>>(w_qkv, wqkvT, 512, 1536);
  wconv_kernel<<<(512 * 512 + 255) / 256, 256, 0, stream>>>(w_proj, wprojT, 512, 512);
  wconv_kernel<<<(512 * 2048 + 255) / 256, 256, 0, stream>>>(w_fc1, wfc1T, 512, 2048);
  wconv_kernel<<<(2048 * 512 + 255) / 256, 256, 0, stream>>>(w_fc2, wfc2T, 2048, 512);

  // 1) LN1 + shift + window partition -> xw (bf16)
  ln1_window_kernel<<<SB_ROWS / 8, 256, 0, stream>>>(hs, ln1_g, ln1_b, xw);

  // 2) QKV GEMM: [131072,512] x [512,1536]
  gemm_bf16_kernel<<<dim3(1536 / 128, SB_ROWS / 128), 256, 0, stream>>>(
      xw, wqkvT, SB_ROWS, 1536, 512, EpiQKV{b_qkv, qkvb});

  // 3) windowed attention (2048*16 units, 4 waves/block)
  attn_kernel<<<(SB_NB * SB_NH) / 4, 128, 0, stream>>>(qkvb, ctxb);

  // 4) proj GEMM + bias + xw residual + window-reverse/unshift + shortcut -> hidden(d_out)
  gemm_bf16_kernel<<<dim3(512 / 128, SB_ROWS / 128), 256, 0, stream>>>(
      ctxb, wprojT, SB_ROWS, 512, 512, EpiProj{b_proj, xw, hs, out});

  // 5) LN2(hidden) -> bf16
  ln2_kernel<<<SB_ROWS / 8, 256, 0, stream>>>(out, ln2_g, ln2_b, ln2buf);

  // 6) MLP, chunked over rows: FC1+GELU -> bf16, FC2 + b + ln2 + hidden -> out
  for (int ch = 0; ch < SB_ROWS / FC_CHUNK; ++ch) {
    const int r0 = ch * FC_CHUNK;
    gemm_bf16_kernel<<<dim3(2048 / 128, FC_CHUNK / 128), 256, 0, stream>>>(
        ln2buf + (size_t)r0 * SB_C, wfc1T, FC_CHUNK, 2048, 512, EpiGelu{b_fc1, fc1buf});
    gemm_bf16_kernel<<<dim3(512 / 128, FC_CHUNK / 128), 256, 0, stream>>>(
        fc1buf, wfc2T, FC_CHUNK, 512, 2048, EpiFC2{b_fc2, ln2buf, out, r0});
  }
}
